// GAT_18116172055064
// MI455X (gfx1250) — matrix operand order repaired and checked by reading, not compiled, not run on hardware
//
#include <hip/hip_runtime.h>
#include <math.h>

// ---------------------------------------------------------------------------
// 2-layer GAT for MI455X (gfx1250).
// GEMMs: bf16 WMMA (v_wmma_f32_16x16x32_bf16), f32 accumulate.
//   - 8 waves/block, each wave owns a 32x64 output tile (8 WMMA accumulators).
//   - B panel staged in LDS (fragment order -> ds_load_b128), double-buffered
//     through registers so global loads overlap WMMA issue.
// Edge softmax/aggregation: f32 with L2 atomics (node arrays fit in 192MB L2).
// ---------------------------------------------------------------------------

typedef __bf16 bf16_t;
typedef bf16_t v8bf  __attribute__((ext_vector_type(8)));
typedef bf16_t v16bf __attribute__((ext_vector_type(16)));
typedef float  v8f   __attribute__((ext_vector_type(8)));

#define NEG_SLOPE 0.2f

__device__ __forceinline__ v8f wmma_bf16(v16bf a, v16bf b, v8f c) {
  return __builtin_amdgcn_wmma_f32_16x16x32_bf16(false, a, false, b, (short)0, c,
                                                 false, false);
}

// Monotone float->uint key so atomicMax(u32) == float max (all finite values).
__device__ __forceinline__ unsigned fkey(float f) {
  unsigned u = __float_as_uint(f);
  return (u & 0x80000000u) ? ~u : (u | 0x80000000u);
}
__device__ __forceinline__ float keyf(unsigned k) {
  unsigned u = (k & 0x80000000u) ? (k & 0x7fffffffu) : ~k;
  return __uint_as_float(u);
}

__device__ __forceinline__ void edge_sd(const int* __restrict__ ei, int E0, int e,
                                        int& s, int& d) {
  if (e < E0) { s = ei[e]; d = ei[E0 + e]; }
  else        { s = e - E0; d = e - E0; }   // self-loops appended
}

// ---- elementwise helpers ---------------------------------------------------
__global__ void k_f32_to_bf16(const float* __restrict__ in, bf16_t* __restrict__ out, int n) {
  int i = blockIdx.x * blockDim.x + threadIdx.x;
  if (i < n) out[i] = (bf16_t)in[i];
}

__global__ void k_fill_u32(unsigned* __restrict__ p, unsigned v, int n) {
  int i = blockIdx.x * blockDim.x + threadIdx.x;
  if (i < n) p[i] = v;
}

__global__ void k_out_bias(float* __restrict__ out, const float* __restrict__ b, int n, int F) {
  int i = blockIdx.x * blockDim.x + threadIdx.x;
  if (i < n) out[i] = b[i % F];
}

// ---- pack W [K,N] f32 (row-major) into WMMA B-fragment order (bf16) --------
// Fragment layout (16-bit B, 32x16 tile): lane l holds col n0+(l&15),
// K values kb + (l>>4)*8 + i (i<8), kb + 16 + (l>>4)*8 + (i-8) (i>=8), stored contiguously per lane.
__global__ void k_pack_w(const float* __restrict__ W, bf16_t* __restrict__ Wp, int K, int N) {
  int idx = blockIdx.x * blockDim.x + threadIdx.x;
  if (idx >= K * N) return;
  int i    = idx & 15;
  int lane = (idx >> 4) & 31;
  int blk  = idx >> 9;          // nt * (K/32) + kbi
  int kT   = K >> 5;
  int kbi  = blk % kT;
  int nt   = blk / kT;
  int col  = nt * 16 + (lane & 15);
  int k    = kbi * 32 + (lane >> 4) * 8 + (i < 8 ? i : 8 + i);
  Wp[idx] = (bf16_t)W[(size_t)k * N + col];
}

// ---- bf16 WMMA GEMM: D[M,N] = A[M,K] * B[K,N] ------------------------------
// Block = 256 threads (8 waves) computes a 256x64 tile; each wave a 32x64 tile
// (two 16-row tiles x four 16-col tiles = 8 accumulators).
// B panel (64 cols x 64 k = 8KB, fragment order) staged in LDS, double-buffered.
// N multiple of 64, K multiple of 64; M rows clamp-guarded (stores uniform per
// 16-row tile since M % 16 == 0). No divergent returns: barriers stay legal and
// EXEC is all-ones at every WMMA.
__global__ void __launch_bounds__(256) k_wmma_gemm(
    const bf16_t* __restrict__ A, const bf16_t* __restrict__ Bp,
    float* __restrict__ D, int M, int N, int K) {
  __shared__ alignas(64) bf16_t lds[2][8 * 32 * 16];   // 2 x 8KB (4096 elems each)

  const int tid  = threadIdx.x;
  const int lane = tid & 31;
  const int wave = tid >> 5;
  const int kT   = K >> 5;
  const int m0   = blockIdx.x * 256 + wave * 32;       // wave's rows: m0..m0+31
  const int n0   = blockIdx.y * 64;
  const int nt0  = n0 >> 4;

  // A addressing (clamped rows keep all lanes active; garbage rows never stored)
  int r0 = m0 + (lane & 15);      if (r0 >= M) r0 = 0;
  int r1 = m0 + 16 + (lane & 15); if (r1 >= M) r1 = 0;
  const int offA = (lane >> 4) * 8;
  const bf16_t* a0p = A + (size_t)r0 * K + offA;
  const bf16_t* a1p = A + (size_t)r1 * K + offA;

  // B staging: 8 sub-blocks of 1KB (t = sub>>1 n-tile, kbl = sub&1 k-block);
  // wave `sub` copies sub-block `sub`, 32B per lane.
  const int sub  = tid >> 5;
  const int s_t  = sub >> 1;
  const int s_kb = sub & 1;
  // Pure addrspace(3) arithmetic (no pointer array!) so stores lower to
  // ds_store_b128, not flat_store.
  bf16_t* const ldsbase = &lds[0][(sub * 32 + lane) * 16];
  auto bsrc = [&](int kc0) {
    int kbi = (kc0 >> 5) + s_kb;
    return Bp + (((size_t)(nt0 + s_t) * kT + kbi) * 32 + lane) * 16;
  };

  v8f acc[8] = {};

  // stage chunk 0 into buffer 0
  *(v16bf*)ldsbase = *(const v16bf*)bsrc(0);
  __syncthreads();

  const int nchunks = K >> 6;                          // 64 k per chunk
  for (int c = 0; c < nchunks; ++c) {
    const int kc0 = c << 6;
    const int cur = c & 1;
    const bool hasNext = (c + 1) < nchunks;            // block-uniform
    v16bf nxt;
    if (hasNext) nxt = *(const v16bf*)bsrc(kc0 + 64);  // overlap with WMMAs below

#pragma unroll
    for (int kbl = 0; kbl < 2; ++kbl) {
      const int kb = kc0 + kbl * 32;
      union { v16bf v; v8bf h[2]; } a0, a1;
      a0.h[0] = *(const v8bf*)(a0p + kb);
      a0.h[1] = *(const v8bf*)(a0p + kb + 16);
      a1.h[0] = *(const v8bf*)(a1p + kb);
      a1.h[1] = *(const v8bf*)(a1p + kb + 16);
#pragma unroll
      for (int t = 0; t < 4; ++t) {
        v16bf b = *(const v16bf*)&lds[cur][((t * 2 + kbl) * 32 + lane) * 16];
        acc[t]     = wmma_bf16(a0.v, b, acc[t]);
        acc[4 + t] = wmma_bf16(a1.v, b, acc[4 + t]);
      }
    }

    if (hasNext) {
      __syncthreads();                                 // everyone done reading buf^1
      *(v16bf*)(ldsbase + ((size_t)(cur ^ 1) << 12)) = nxt;   // 4096 elems/buffer
      __syncthreads();                                 // buf^1 visible to all
    }
  }

  // C/D layout: VGPR r -> M = r + (lane<16 ? 0 : 8), N = n0 + t*16 + (lane&15)
  const int colBase = n0 + (lane & 15);
  if (m0 < M) {
    const int rbase = m0 + (lane >> 4) * 8;
#pragma unroll
    for (int t = 0; t < 4; ++t)
#pragma unroll
      for (int r = 0; r < 8; ++r)
        D[(size_t)(rbase + r) * N + colBase + t * 16] = acc[t][r];
  }
  if (m0 + 16 < M) {
    const int rbase = m0 + 16 + (lane >> 4) * 8;
#pragma unroll
    for (int t = 0; t < 4; ++t)
#pragma unroll
      for (int r = 0; r < 8; ++r)
        D[(size_t)(rbase + r) * N + colBase + t * 16] = acc[4 + t][r];
  }
}

// ---- per-node attention logits: al[n,h] = dot(h[n,h,:], a[h,:]) ------------
__global__ void k_node_logits(const float* __restrict__ h, const float* __restrict__ as_,
                              const float* __restrict__ ad_, float* __restrict__ als,
                              float* __restrict__ ald, int n_nodes, int heads, int ch) {
  int idx = blockIdx.x * blockDim.x + threadIdx.x;
  if (idx >= n_nodes * heads) return;
  int hd = idx % heads;
  int n  = idx / heads;
  const float* hp = h + (size_t)n * heads * ch + (size_t)hd * ch;
  float ss = 0.f, sd = 0.f;
  for (int c = 0; c < ch; ++c) {
    float v = hp[c];
    ss = fmaf(v, as_[hd * ch + c], ss);
    sd = fmaf(v, ad_[hd * ch + c], sd);
  }
  als[idx] = ss;
  ald[idx] = sd;
}

// ---- edge pass A: e = leaky_relu(al_s[src]+al_d[dst]); segment max ---------
__global__ void k_edge_max(const int* __restrict__ ei, int E0, int E, int heads,
                           const float* __restrict__ als, const float* __restrict__ ald,
                           float* __restrict__ ebuf, unsigned* __restrict__ mkey) {
  int idx = blockIdx.x * blockDim.x + threadIdx.x;
  if (idx >= E * heads) return;
  int h = idx % heads, e = idx / heads;
  int s, d; edge_sd(ei, E0, e, s, d);
  float v = als[s * heads + h] + ald[d * heads + h];
  v = v > 0.f ? v : NEG_SLOPE * v;
  ebuf[idx] = v;
  atomicMax(&mkey[d * heads + h], fkey(v));
}

// ---- edge pass B: p = exp(e - m[dst]); segment sum -------------------------
__global__ void k_edge_exp(const int* __restrict__ ei, int E0, int E, int heads,
                           float* __restrict__ ebuf, const unsigned* __restrict__ mkey,
                           float* __restrict__ ssum) {
  int idx = blockIdx.x * blockDim.x + threadIdx.x;
  if (idx >= E * heads) return;
  int h = idx % heads, e = idx / heads;
  int s, d; edge_sd(ei, E0, e, s, d);
  (void)s;
  float p = expf(ebuf[idx] - keyf(mkey[d * heads + h]));
  ebuf[idx] = p;
  atomicAdd(&ssum[d * heads + h], p);
}

// ---- edge pass C: out[dst] += (p/s[dst]) * h[src] --------------------------
// One 256-thread block per edge: src/dst fetched once, channels coalesced,
// head index from one integer division (c / ch) per channel.
__global__ void k_edge_agg(const int* __restrict__ ei, int E0, int heads, int ch,
                           const float* __restrict__ ebuf, const float* __restrict__ ssum,
                           const float* __restrict__ hf, float* __restrict__ out) {
  const int e = blockIdx.x;
  const int F = heads * ch;
  int s, d; edge_sd(ei, E0, e, s, d);
  for (int c = threadIdx.x; c < F; c += blockDim.x) {
    int h = c / ch;
    float alpha = ebuf[(size_t)e * heads + h] / ssum[(size_t)d * heads + h];
    atomicAdd(&out[(size_t)d * F + c], alpha * hf[(size_t)s * F + c]);
  }
}

// ---- bias + ELU + downconvert to bf16 (layer-2 GEMM input) -----------------
__global__ void k_bias_elu_bf16(const float* __restrict__ agg, const float* __restrict__ bias,
                                bf16_t* __restrict__ out, int n, int F) {
  int idx = blockIdx.x * blockDim.x + threadIdx.x;
  if (idx >= n) return;
  int c = idx % F;
  float v = agg[idx] + bias[c];
  v = v > 0.f ? v : (expf(v) - 1.f);
  out[idx] = (bf16_t)v;
}

// ---------------------------------------------------------------------------
extern "C" void kernel_launch(void* const* d_in, const int* in_sizes, int n_in,
                              void* d_out, int out_size, void* d_ws, size_t ws_size,
                              hipStream_t stream) {
  (void)in_sizes; (void)n_in; (void)out_size; (void)ws_size;
  const float* x   = (const float*)d_in[0];
  const int*   ei  = (const int*)  d_in[1];
  const float* W1  = (const float*)d_in[2];
  const float* a1s = (const float*)d_in[3];
  const float* a1d = (const float*)d_in[4];
  const float* b1  = (const float*)d_in[5];
  const float* W2  = (const float*)d_in[6];
  const float* a2s = (const float*)d_in[7];
  const float* a2d = (const float*)d_in[8];
  const float* b2  = (const float*)d_in[9];
  float* out = (float*)d_out;

  const int NN = 30000, E0 = 480000, EE = E0 + NN;   // self-loops appended
  const int K1 = 1024, F1 = 512, H1 = 8, C1 = 64;
  const int K2 = 512,  F2 = 256;

  // workspace layout (layer 2 aliases dead layer-1 buffers); peak ~206 MB
  char* ws = (char*)d_ws;
  size_t off = 0;
  auto alloc = [&](size_t bytes) -> char* {
    char* p = ws + off;
    off += (bytes + 255) & ~(size_t)255;
    return p;
  };
  bf16_t*   xbf  = (bf16_t*)  alloc((size_t)NN * K1 * 2);   // 61.4 MB
  bf16_t*   w1p  = (bf16_t*)  alloc((size_t)K1 * F1 * 2);   //  1.0 MB
  float*    h1   = (float*)   alloc((size_t)NN * F1 * 4);   // 61.4 MB
  float*    al1s = (float*)   alloc((size_t)NN * H1 * 4);
  float*    al1d = (float*)   alloc((size_t)NN * H1 * 4);
  unsigned* m1   = (unsigned*)alloc((size_t)NN * H1 * 4);
  float*    s1   = (float*)   alloc((size_t)NN * H1 * 4);
  float*    e1   = (float*)   alloc((size_t)EE * H1 * 4);   // 16.3 MB
  float*    agg1 = (float*)   alloc((size_t)NN * F1 * 4);   // 61.4 MB
  // layer-2 aliases (all strictly smaller than the aliased region)
  bf16_t*   h2bf = xbf;
  bf16_t*   w2p  = w1p;
  float*    h2   = h1;
  float*    al2s = al1s;  float* al2d = al1d;
  unsigned* m2   = m1;    float* s2   = s1;
  float*    e2   = e1;

  const int T = 256;
  auto nb = [](long long n, int t) { return (unsigned)((n + t - 1) / t); };
  const unsigned mBlk = (unsigned)((NN + 255) / 256);       // 118 (256 rows/block)

  // ---------------- layer 1 ----------------
  k_f32_to_bf16<<<nb((long long)NN * K1, T), T, 0, stream>>>(x, xbf, NN * K1);
  k_pack_w     <<<nb(K1 * F1, T), T, 0, stream>>>(W1, w1p, K1, F1);
  k_wmma_gemm  <<<dim3(mBlk, F1 / 64), T, 0, stream>>>(xbf, w1p, h1, NN, F1, K1);
  k_node_logits<<<nb(NN * H1, T), T, 0, stream>>>(h1, a1s, a1d, al1s, al1d, NN, H1, C1);

  k_fill_u32<<<nb(NN * H1, T), T, 0, stream>>>(m1, 0u, NN * H1);             // key(-inf-ish)
  k_fill_u32<<<nb(NN * H1, T), T, 0, stream>>>((unsigned*)s1, 0u, NN * H1);
  k_fill_u32<<<nb((long long)NN * F1, T), T, 0, stream>>>((unsigned*)agg1, 0u, NN * F1);

  k_edge_max<<<nb((long long)EE * H1, T), T, 0, stream>>>(ei, E0, EE, H1, al1s, al1d, e1, m1);
  k_edge_exp<<<nb((long long)EE * H1, T), T, 0, stream>>>(ei, E0, EE, H1, e1, m1, s1);
  k_edge_agg<<<EE, T, 0, stream>>>(ei, E0, H1, C1, e1, s1, h1, agg1);

  k_bias_elu_bf16<<<nb((long long)NN * F1, T), T, 0, stream>>>(agg1, b1, h2bf, NN * F1, F1);

  // ---------------- layer 2 ----------------
  k_pack_w     <<<nb(K2 * F2, T), T, 0, stream>>>(W2, w2p, K2, F2);
  k_wmma_gemm  <<<dim3(mBlk, F2 / 64), T, 0, stream>>>(h2bf, w2p, h2, NN, F2, K2);
  k_node_logits<<<nb(NN, T), T, 0, stream>>>(h2, a2s, a2d, al2s, al2d, NN, 1, F2);

  k_fill_u32<<<nb(NN, T), T, 0, stream>>>(m2, 0u, NN);
  k_fill_u32<<<nb(NN, T), T, 0, stream>>>((unsigned*)s2, 0u, NN);

  k_edge_max<<<nb(EE, T), T, 0, stream>>>(ei, E0, EE, 1, al2s, al2d, e2, m2);
  k_edge_exp<<<nb(EE, T), T, 0, stream>>>(ei, E0, EE, 1, e2, m2, s2);

  k_out_bias<<<nb((long long)NN * F2, T), T, 0, stream>>>(out, b2, NN * F2, F2);
  k_edge_agg<<<EE, T, 0, stream>>>(ei, E0, 1, F2, e2, s2, h2, out);
}
